// CausalTemporalAttention_42691974922472
// MI455X (gfx1250) — matrix operand order repaired and checked by reading, not compiled, not run on hardware
//
#include <hip/hip_runtime.h>

// ---------------------------------------------------------------------------
// CDNA5 (gfx1250, wave32) causal temporal attention.
//  - all matrix math through v_wmma_f32_16x16x32_bf16 (fp32 accumulate)
//  - GEMM A-panels staged to LDS via global_load_async_to_lds_b128 (ASYNCcnt)
//  - attention mean computed atomic-free by a stats-replay WMMA kernel
// ---------------------------------------------------------------------------

typedef __attribute__((ext_vector_type(16))) __bf16          v16bf;
typedef __attribute__((ext_vector_type(8)))  float           v8f;
typedef __attribute__((ext_vector_type(16))) unsigned short  v16u;
typedef __attribute__((ext_vector_type(8)))  unsigned short  v8u;
typedef int v4i __attribute__((vector_size(16)));            // matches builtin param

#define AS1_ __attribute__((address_space(1)))
#define AS3_ __attribute__((address_space(3)))

#define B_    2
#define S_    2048
#define H_    1024
#define NH_   16
#define HD_   64
#define H3_   3072
#define BS_   4096          // B_*S_
#define SCALE_ 0.125f       // hd^-0.5
#define QTILES_ 128         // S_/16

#if defined(__gfx1250__) && __has_builtin(__builtin_amdgcn_global_load_async_to_lds_b128) && \
    __has_builtin(__builtin_amdgcn_s_wait_asynccnt)
#define ASYNC_LDS 1
#else
#define ASYNC_LDS 0
#endif

__device__ __forceinline__ unsigned short f32_to_bf16(float f) {
  unsigned int u = __float_as_uint(f);
  u = (u + 0x7FFFu + ((u >> 16) & 1u)) >> 16;   // round-to-nearest-even
  return (unsigned short)u;
}

__device__ __forceinline__ v8f bfwmma(v16u a, v16u b, v8f c) {
  return __builtin_amdgcn_wmma_f32_16x16x32_bf16(
      false, __builtin_bit_cast(v16bf, a),
      false, __builtin_bit_cast(v16bf, b),
      (short)0, c, false, false);
}

// Gather a bf16 A-fragment (16x32, M x K) from a row-major [*, rowStride] matrix.
// lane<16: M=lane, K=0..15; lane>=16: M=lane-16, K=16..31.
__device__ __forceinline__ v16u load_afrag(const unsigned short* base,
                                           size_t rowStride, int m0, int k0, int lane) {
  const int m    = m0 + (lane & 15);
  const int koff = (lane < 16) ? 0 : 16;
  const unsigned short* p = base + (size_t)m * rowStride + k0 + koff;
  v8u lo = *(const v8u*)p;
  v8u hi = *(const v8u*)(p + 8);
  v16u u;
#pragma unroll
  for (int e = 0; e < 8; ++e) { u[e] = lo[e]; u[e + 8] = hi[e]; }
  return u;
}

// Stage a 16 x 1024 bf16 row panel (rows m0..m0+15 of a [*,1024] matrix) into
// LDS, using the CDNA5 async global->LDS DMA path when available.
__device__ __forceinline__ void stage_panel(const unsigned short* __restrict__ g,
                                            int m0, unsigned short* panel) {
  const int t = threadIdx.x;                    // 128 threads
#if ASYNC_LDS
#pragma unroll
  for (int i = 0; i < 16; ++i) {
    const int chunk = i * 128 + t;              // 16-byte chunks, 2048 total
    const int row = chunk >> 7;                 // 128 chunks per 1024-elem row
    const int col = (chunk & 127) * 8;          // ushort offset
    __builtin_amdgcn_global_load_async_to_lds_b128(
        (AS1_ v4i*)(g + (size_t)(m0 + row) * H_ + col),
        (AS3_ v4i*)(panel + (size_t)row * H_ + col),
        0, 0);
  }
  __builtin_amdgcn_s_wait_asynccnt(0);
  __syncthreads();
#else
#pragma unroll
  for (int i = 0; i < 16; ++i) {
    const int chunk = i * 128 + t;
    const int row = chunk >> 7;
    const int col = (chunk & 127) * 8;
    *(v8u*)(panel + (size_t)row * H_ + col) =
        *(const v8u*)(g + (size_t)(m0 + row) * H_ + col);
  }
  __syncthreads();
#endif
}

// A-fragment gather out of an LDS panel (ds_load_b128 pairs)
__device__ __forceinline__ v16u afrag_lds(const unsigned short* panel, int k0, int lane) {
  const int m    = lane & 15;
  const int koff = (lane < 16) ? 0 : 16;
  const unsigned short* p = panel + (size_t)m * H_ + k0 + koff;
  v8u lo = *(const v8u*)p;
  v8u hi = *(const v8u*)(p + 8);
  v16u u;
#pragma unroll
  for (int e = 0; e < 8; ++e) { u[e] = lo[e]; u[e + 8] = hi[e]; }
  return u;
}

// ------------------------------- conversions -------------------------------

__global__ void k_cvt(const float* __restrict__ src, unsigned short* __restrict__ dst, int n) {
  int i = blockIdx.x * blockDim.x + threadIdx.x;
  if (i < n) dst[i] = f32_to_bf16(src[i]);
}

// w[rows][cols] (f32) -> wt[cols][rows] (bf16)
__global__ void k_cvt_t(const float* __restrict__ w, unsigned short* __restrict__ wt,
                        int rows, int cols) {
  int i = blockIdx.x * blockDim.x + threadIdx.x;
  if (i < rows * cols) {
    int r = i / cols, c = i % cols;
    wt[(size_t)c * rows + r] = f32_to_bf16(w[i]);
  }
}

__global__ void k_zero(float* __restrict__ p, int n) {
  int i = blockIdx.x * blockDim.x + threadIdx.x;
  if (i < n) p[i] = 0.0f;
}

// ------------------------------- QKV GEMM ----------------------------------
// qkv = x(bf16)[4096x1024] @ w_qkvT(bf16)[3072x1024]^T ; scatter to
// Q,K : [b][h][s][d] bf16      V : [b][h][d][s] bf16 (transposed for PV)

__global__ void k_qkv(const unsigned short* __restrict__ xb,
                      const unsigned short* __restrict__ wqkvT,
                      unsigned short* __restrict__ Qb,
                      unsigned short* __restrict__ Kb,
                      unsigned short* __restrict__ Vt) {
  __shared__ unsigned short panelA[16 * H_];    // 32KB, shared by 4 waves
  const int lane = threadIdx.x & 31;
  const int wave = blockIdx.x * 4 + (threadIdx.x >> 5);
  const int tilesN = H3_ / 16;                  // 192 (mult of 4 -> same m0/block)
  const int m0 = (wave / tilesN) * 16;
  const int n0 = (wave % tilesN) * 16;

  stage_panel(xb, m0, panelA);

  const int nrow = n0 + (lane & 15);            // B-fragment column (output col)
  const int kb   = (lane < 16) ? 0 : 16;

  v8f acc = {0.f, 0.f, 0.f, 0.f, 0.f, 0.f, 0.f, 0.f};
  for (int k0 = 0; k0 < H_; k0 += 32) {
    v16u a = afrag_lds(panelA, k0, lane);
    v16u b = *(const v16u*)(wqkvT + (size_t)nrow * H_ + k0 + kb);
    acc = bfwmma(a, b, acc);
  }

  // scatter: one column per lane; 16 cols of a tile share (section, head)
  const int col = n0 + (lane & 15);
  const int sec = col >> 10, cin = col & 1023;
  const int h = cin >> 6, d = cin & 63;
  const int rbase = m0 + ((lane < 16) ? 0 : 8);
#pragma unroll
  for (int r = 0; r < 8; ++r) {
    const int gr = rbase + r;
    const int b_ = gr >> 11, s = gr & (S_ - 1);
    const unsigned short v = f32_to_bf16(acc[r]);
    const size_t bh = (size_t)(b_ * NH_ + h);
    if (sec == 0)      Qb[(bh * S_ + s) * HD_ + d] = v;
    else if (sec == 1) Kb[(bh * S_ + s) * HD_ + d] = v;
    else               Vt[(bh * HD_ + d) * S_ + s] = v;
  }
}

// ------------------------------ attention ----------------------------------
// One wave per (b, h, 16-query tile). Softmax over the FULL row (the mask is
// applied after softmax per the reference). Pass 1 computes (m, 1/l) stats and
// saves them for k_mean; pass 2 does masked P@V. No atomics anywhere.

__device__ __forceinline__ v8f score_tile(v16u aQ0, v16u aQ1,
                                          const unsigned short* Kh, int kt, int lane) {
  const int j  = kt * 16 + (lane & 15);
  const int kb = (lane < 16) ? 0 : 16;
  const unsigned short* p = Kh + (size_t)j * HD_ + kb;
  v16u b0 = *(const v16u*)p;          // d = 0..31 chunk
  v16u b1 = *(const v16u*)(p + 32);   // d = 32..63 chunk
  v8f c = {0.f, 0.f, 0.f, 0.f, 0.f, 0.f, 0.f, 0.f};
  c = bfwmma(aQ0, b0, c);
  c = bfwmma(aQ1, b1, c);
  return c;
}

__global__ void k_attn(const unsigned short* __restrict__ Qb,
                       const unsigned short* __restrict__ Kb,
                       const unsigned short* __restrict__ Vt,
                       unsigned short* __restrict__ aout,
                       float2* __restrict__ stats) {
  __shared__ float lds_p[4][16 * 32];          // per-wave P tile, 2KB each
  const int lane = threadIdx.x & 31;
  const int wid  = threadIdx.x >> 5;
  const int unit = blockIdx.x * 4 + wid;       // b*NH*QTILES + h*QTILES + qt
  const int qt = unit & (QTILES_ - 1);
  const int h  = (unit >> 7) & (NH_ - 1);
  const int b  = unit >> 11;

  const size_t bh = (size_t)(b * NH_ + h);
  const unsigned short* Qh = Qb + bh * S_ * HD_;
  const unsigned short* Kh = Kb + bh * S_ * HD_;
  const unsigned short* Vh = Vt + bh * HD_ * S_;

  // Q A-fragments for the two 32-wide d-chunks
  v16u aQ0 = load_afrag(Qh, HD_, qt * 16, 0,  lane);
  v16u aQ1 = load_afrag(Qh, HD_, qt * 16, 32, lane);

  // ---- pass 1: online softmax stats over all 128 key tiles ----
  float m[8], l[8];
#pragma unroll
  for (int r = 0; r < 8; ++r) { m[r] = -3.0e38f; l[r] = 0.0f; }

  for (int kt = 0; kt < QTILES_; ++kt) {
    v8f s = score_tile(aQ0, aQ1, Kh, kt, lane);
#pragma unroll
    for (int r = 0; r < 8; ++r) {
      float v = s[r] * SCALE_;
      float rmax = v;                                   // reduce over 16 cols
      rmax = fmaxf(rmax, __shfl_xor(rmax, 1, 32));
      rmax = fmaxf(rmax, __shfl_xor(rmax, 2, 32));
      rmax = fmaxf(rmax, __shfl_xor(rmax, 4, 32));
      rmax = fmaxf(rmax, __shfl_xor(rmax, 8, 32));
      float mn = fmaxf(m[r], rmax);
      float ex = __expf(v - mn);
      ex += __shfl_xor(ex, 1, 32);
      ex += __shfl_xor(ex, 2, 32);
      ex += __shfl_xor(ex, 4, 32);
      ex += __shfl_xor(ex, 8, 32);
      l[r] = l[r] * __expf(m[r] - mn) + ex;
      m[r] = mn;
    }
  }
  float inv[8];
#pragma unroll
  for (int r = 0; r < 8; ++r) inv[r] = 1.0f / l[r];

  // save per-row stats (m, 1/l) for the mean-replay kernel; every lane in a
  // 16-lane half holds identical reduced values -> lanes 0 and 16 store.
  if ((lane & 15) == 0) {
    const int rb = (lane < 16) ? 0 : 8;
#pragma unroll
    for (int r = 0; r < 8; ++r)
      stats[bh * S_ + qt * 16 + rb + r] = make_float2(m[r], inv[r]);
  }

  // ---- pass 2: masked tiles only (mask uniform per tile: kt/4 <= qt/4) ----
  v8f o[4];
#pragma unroll
  for (int nt = 0; nt < 4; ++nt) o[nt] = (v8f){0.f,0.f,0.f,0.f,0.f,0.f,0.f,0.f};

  float* pl = &lds_p[wid][0];
  const int nkt = ((qt >> 2) + 1) * 4;         // allowed key tiles (mult of 4)
  const int koff = (lane < 16) ? 0 : 8;
  const int kb2  = (lane < 16) ? 0 : 16;

  for (int kt2 = 0; kt2 < nkt; kt2 += 2) {     // 32 keys per step
#pragma unroll
    for (int t = 0; t < 2; ++t) {
      v8f s = score_tile(aQ0, aQ1, Kh, kt2 + t, lane);
#pragma unroll
      for (int r = 0; r < 8; ++r) {
        const int row = r + ((lane < 16) ? 0 : 8);
        float p = __expf(s[r] * SCALE_ - m[r]) * inv[r];
        pl[row * 32 + t * 16 + (lane & 15)] = p;
      }
    }
    // wave-private LDS slab; LDS is in-order per wave — fence the compiler
    asm volatile("s_wait_dscnt 0" ::: "memory");

    // P as A-fragment (16x32), bf16
    v16u aP;
    const int mr = lane & 15;
#pragma unroll
    for (int e = 0; e < 8; ++e) {
      aP[e]     = f32_to_bf16(pl[mr * 32 + 2 * koff + e]);
      aP[e + 8] = f32_to_bf16(pl[mr * 32 + 2 * koff + 8 + e]);
    }
    // V B-fragments: B[k][n] = Vt[d0+n][j0+k] (contiguous in k)
#pragma unroll
    for (int nt = 0; nt < 4; ++nt) {
      const int d = nt * 16 + (lane & 15);
      v16u bV = *(const v16u*)(Vh + (size_t)d * S_ + kt2 * 16 + kb2);
      o[nt] = bfwmma(aP, bV, o[nt]);
    }
  }

  // store per-head output as bf16 rows of [BS][H]
  unsigned short* ao = aout + ((size_t)b * S_ + qt * 16) * H_ + h * HD_;
#pragma unroll
  for (int nt = 0; nt < 4; ++nt)
#pragma unroll
    for (int r = 0; r < 8; ++r) {
      const int row = r + ((lane < 16) ? 0 : 8);
      ao[(size_t)row * H_ + nt * 16 + (lane & 15)] = f32_to_bf16(o[nt][r]);
    }
}

// --------------------------- attention-mean replay -------------------------
// One wave per (b, qt, jt) unmasked tile: recompute QK^T per head with WMMA,
// apply saved softmax stats, average over heads, plain-store (no atomics).

__global__ void k_mean(const unsigned short* __restrict__ Qb,
                       const unsigned short* __restrict__ Kb,
                       const float2* __restrict__ stats,
                       float* __restrict__ mean) {
  const int lane = threadIdx.x & 31;
  const int wid  = threadIdx.x >> 5;
  const int bq = blockIdx.x;                    // b*128 + qt
  const int qt = bq & (QTILES_ - 1);
  const int b  = bq >> 7;
  const int jt = blockIdx.y * 4 + wid;
  if (jt > (qt | 3)) return;                    // block-causal: jt/4 <= qt/4

  float macc[8] = {0.f, 0.f, 0.f, 0.f, 0.f, 0.f, 0.f, 0.f};
  const int rbase = (lane < 16) ? 0 : 8;

  for (int h = 0; h < NH_; ++h) {
    const size_t bh = (size_t)(b * NH_ + h);
    const unsigned short* Qh = Qb + bh * S_ * HD_;
    const unsigned short* Kh = Kb + bh * S_ * HD_;
    v16u aQ0 = load_afrag(Qh, HD_, qt * 16, 0,  lane);
    v16u aQ1 = load_afrag(Qh, HD_, qt * 16, 32, lane);
    v8f s = score_tile(aQ0, aQ1, Kh, jt, lane);
#pragma unroll
    for (int r = 0; r < 8; ++r) {
      const float2 st = stats[bh * S_ + qt * 16 + rbase + r];
      macc[r] += __expf(s[r] * SCALE_ - st.x) * st.y;
    }
  }
#pragma unroll
  for (int r = 0; r < 8; ++r) {
    const int row = qt * 16 + rbase + r;
    mean[((size_t)b * S_ + row) * S_ + jt * 16 + (lane & 15)] = macc[r] * (1.0f / NH_);
  }
}

// ------------------------------ out projection -----------------------------

__global__ void k_out(const unsigned short* __restrict__ aoutb,
                      const unsigned short* __restrict__ woutT,
                      const float* __restrict__ bias,
                      float* __restrict__ out) {
  __shared__ unsigned short panelA[16 * H_];    // 32KB, shared by 4 waves
  const int lane = threadIdx.x & 31;
  const int wave = blockIdx.x * 4 + (threadIdx.x >> 5);
  const int tilesN = H_ / 16;                   // 64 (mult of 4 -> same m0/block)
  const int m0 = (wave / tilesN) * 16;
  const int n0 = (wave % tilesN) * 16;

  stage_panel(aoutb, m0, panelA);

  const int nrow = n0 + (lane & 15);
  const int kb   = (lane < 16) ? 0 : 16;

  v8f acc = {0.f, 0.f, 0.f, 0.f, 0.f, 0.f, 0.f, 0.f};
  for (int k0 = 0; k0 < H_; k0 += 32) {
    v16u a = afrag_lds(panelA, k0, lane);
    v16u b = *(const v16u*)(woutT + (size_t)nrow * H_ + k0 + kb);
    acc = bfwmma(a, b, acc);
  }
  const float bv = bias[n0 + (lane & 15)];
  const int rbase = m0 + ((lane < 16) ? 0 : 8);
#pragma unroll
  for (int r = 0; r < 8; ++r)
    out[(size_t)(rbase + r) * H_ + n0 + (lane & 15)] = acc[r] + bv;
}

// ------------------------------- launcher ----------------------------------

extern "C" void kernel_launch(void* const* d_in, const int* in_sizes, int n_in,
                              void* d_out, int out_size, void* d_ws, size_t ws_size,
                              hipStream_t stream) {
  const float* x     = (const float*)d_in[0];   // [2,2048,1024]
  const float* wqkv  = (const float*)d_in[1];   // [1024,3072]
  const float* wout  = (const float*)d_in[2];   // [1024,1024]
  const float* bout  = (const float*)d_in[3];   // [1024]
  float* out = (float*)d_out;                   // [4.19M output][8.39M attn-mean]

  unsigned short* xb    = (unsigned short*)d_ws;
  unsigned short* wqkvT = xb    + (size_t)BS_ * H_;
  unsigned short* woutT = wqkvT + (size_t)H_ * H3_;
  unsigned short* Qb    = woutT + (size_t)H_ * H_;
  unsigned short* Kb    = Qb    + (size_t)BS_ * H_;
  unsigned short* Vt    = Kb    + (size_t)BS_ * H_;
  unsigned short* aoutb = Vt    + (size_t)BS_ * H_;
  float2*         stats = (float2*)(aoutb + (size_t)BS_ * H_);  // B*NH*S float2

  float* mean = out + (size_t)BS_ * H_;

  const int nx = BS_ * H_;
  k_cvt  <<<(nx + 255) / 256, 256, 0, stream>>>(x, xb, nx);
  k_cvt_t<<<(H_ * H3_ + 255) / 256, 256, 0, stream>>>(wqkv, wqkvT, H_, H3_);
  k_cvt_t<<<(H_ * H_  + 255) / 256, 256, 0, stream>>>(wout, woutT, H_, H_);
  const int nm = B_ * S_ * S_;
  k_zero <<<(nm + 255) / 256, 256, 0, stream>>>(mean, nm);   // masked region = 0

  // QKV: (4096/16)*(3072/16) = 49152 wave-tiles, 4 waves/block
  k_qkv <<<49152 / 4, 128, 0, stream>>>(xb, wqkvT, Qb, Kb, Vt);
  // attention: B*NH*(S/16) = 4096 wave-units, 4 waves/block
  k_attn<<<4096 / 4, 128, 0, stream>>>(Qb, Kb, Vt, aoutb, stats);
  // attention-mean replay: x = (b,qt), y = jt/4; masked waves early-exit
  dim3 gmean(B_ * QTILES_, QTILES_ / 4);
  k_mean<<<gmean, 128, 0, stream>>>(Qb, Kb, stats, mean);
  // out-proj: (4096/16)*(1024/16) = 16384 wave-tiles
  k_out <<<16384 / 4, 128, 0, stream>>>(aoutb, woutT, bout, out);
}
